// LearnableTensorSquare_1030792151277
// MI455X (gfx1250) — matrix operand order repaired and checked by reading, not compiled, not run on hardware
//
#include <hip/hip_runtime.h>
#include <hip/hip_fp16.h>

typedef __attribute__((ext_vector_type(16))) _Float16 v16h;
typedef __attribute__((ext_vector_type(8)))  _Float16 v8h;
typedef __attribute__((ext_vector_type(8)))  float    v8f;

#define Z_TOTAL 16384
#define N_IN    127
#define NN      128          // N = N_IN + 1
#define K_OUT   128
#define DIM_TS  2080         // 65 * 32
#define NN2     (NN * NN)    // 16384

// ---------------------------------------------------------------------------
// Stage 0: one-time W f32 -> f16 conversion.
// ---------------------------------------------------------------------------
__global__ __launch_bounds__(256)
void cvt_w_f16(const float* __restrict__ W, _Float16* __restrict__ W16)
{
    const int idx = blockIdx.x * 256 + threadIdx.x;
    if (idx < K_OUT * DIM_TS)
        W16[idx] = (_Float16)W[idx];
}

// ---------------------------------------------------------------------------
// Stage 1:  P[kc, c] = sum_d W[kc, d] * T[d, c]   (f16 in, f16 out, f32 acc)
//   Block = 8 waves = 2 c-tiles x 4 kc-groups (2 kc-tiles each); waves that
//   share a c-tile share the T stream through WGP$ -> T read from HBM once,
//   4096 waves of latency hiding.  Branch-free steady-state software
//   pipeline: all loads for chunk n issue during chunk n-1.
// ---------------------------------------------------------------------------
__global__ __launch_bounds__(256)
void wt_gemm_f16(const _Float16* __restrict__ W16,
                 const float*    __restrict__ T,
                 _Float16*       __restrict__ P)
{
    const int lane = threadIdx.x & 31;
    const int wave = threadIdx.x >> 5;
    const int g    = lane >> 4;              // half-wave group (0/1)
    const int r    = lane & 15;              // A row-in-tile / B col-in-tile
    const int ct   = wave >> 2;              // c-tile within block (0/1)
    const int mg   = wave & 3;               // kc group: tiles mg*2, mg*2+1
    const int nt   = blockIdx.x * 2 + ct;    // 0..1023
    const int c0   = nt * 16;
    const int m0   = mg * 2;

    const float* tbase = T + c0 + r + (size_t)(g * 16) * NN2;

    v8f acc[2] = {};

    // raw-f32 gather of one 32-d chunk of T (16 strided dwords / lane).
    // Prefetch 2 chunks ahead unconditionally: speculative prefetch past the
    // end of T is silently dropped (ISA 10.5), and this keeps the body
    // branch-free so waitcnt insertion stays exact.
    auto loadRaw = [&](int dc, float tb[16]) {
        const float* tcol = tbase + (size_t)dc * NN2;
        #pragma unroll
        for (int e = 0; e < 16; ++e)
            tb[e] = tcol[(size_t)e * NN2];
        __builtin_prefetch(tcol + (size_t)64 * NN2, 0, 1);
    };

    // A fragments (2 kc tiles) for one chunk
    auto loadA = [&](int dc, v16h af[2]) {
        #pragma unroll
        for (int mi = 0; mi < 2; ++mi) {
            const _Float16* wrow =
                W16 + (size_t)((m0 + mi) * 16 + r) * DIM_TS + dc + g * 16;
            const v8h alo = *(const v8h*)wrow;         // k = g*16   + 0..7
            const v8h ahi = *(const v8h*)(wrow + 8);   // k = g*16+8 + 0..7
            af[mi] = __builtin_shufflevector(
                alo, ahi, 0,1,2,3,4,5,6,7, 8,9,10,11,12,13,14,15);
        }
    };

    // cvt + 2 WMMAs for one chunk (all inputs were loaded a chunk earlier)
    auto doWmma = [&](const float tb[16], const v16h af[2], v16h& b) {
        #pragma unroll
        for (int e = 0; e < 16; ++e) b[e] = (_Float16)tb[e];
        #pragma unroll
        for (int mi = 0; mi < 2; ++mi)
            acc[mi] = __builtin_amdgcn_wmma_f32_16x16x32_f16(
                          false, af[mi], false, b, (short)0, acc[mi],
                          false, false);
    };

    float tb0[16], tb1[16];
    v16h  af0[2], af1[2];
    v16h  bc0, bc1;                          // per-parity cvt targets

    loadRaw(0, tb0);
    loadA(0, af0);
    // steady state: consume chunks 0..63, always preload it+1 and it+2 (<=64)
    for (int it = 0; it < 64; it += 2) {
        loadRaw((it + 1) * 32, tb1);
        loadA((it + 1) * 32, af1);
        doWmma(tb0, af0, bc0);
        loadRaw((it + 2) * 32, tb0);
        loadA((it + 2) * 32, af0);
        doWmma(tb1, af1, bc1);
    }
    // epilogue: chunk 64 already resident in tb0/af0
    doWmma(tb0, af0, bc0);

    // ---- store C/D layout: element e -> row (e + g*8), col r
    #pragma unroll
    for (int mi = 0; mi < 2; ++mi)
        #pragma unroll
        for (int e = 0; e < 8; ++e)
            P[(size_t)((m0 + mi) * 16 + e + g * 8) * NN2 + c0 + r]
                = (_Float16)acc[mi][e];
}

// ---------------------------------------------------------------------------
// Stage 2:  out[z, k] = sum_ij P[k, ij] * f[z,i] * f[z,j]
//   G built on the fly from an f16 LDS f-tile (8 v_pk_mul_f16 per fragment).
//   Block = 64 z x 128 k; wave owns 4 k-tiles.  Branch-free double-buffered
//   steady state over the 512 inner chunks, peeled epilogue.
// ---------------------------------------------------------------------------
__global__ __launch_bounds__(256)
void tsq_gemm_f16(const float*    __restrict__ feat,
                  const _Float16* __restrict__ P,
                  float*          __restrict__ out)
{
    // pitch 136 halfs = 68 dwords -> per-row bank offset 4 (mod 64): conflict-free
    __shared__ _Float16 flsh[64][NN + 8];

    const int tid = threadIdx.x;
    const int z0b = blockIdx.x * 64;

    for (int idx = tid; idx < 64 * NN; idx += 256) {
        const int rr = idx >> 7;
        const int cc = idx & 127;
        flsh[rr][cc] = (cc == 0)
                         ? (_Float16)1.0f
                         : (_Float16)feat[(size_t)(z0b + rr) * N_IN + cc - 1];
    }
    __syncthreads();

    const int lane = tid & 31;
    const int wave = tid >> 5;
    const int g    = lane >> 4;
    const int r    = lane & 15;
    const int zt   = wave & 3;               // 16-row z sub-tile
    const int kg   = wave >> 2;              // k half: columns kg*64 .. +63
    const int zr   = zt * 16 + r;
    const int z0   = z0b + zt * 16;

    v8f acc[4] = {};

    auto loadB = [&](int chunk, v16h b[4]) {
        const int ij = chunk * 32 + g * 16;
        #pragma unroll
        for (int q = 0; q < 4; ++q)
            b[q] = *(const v16h*)(P + (size_t)(kg * 64 + q * 16 + r) * NN2 + ij);
    };
    auto makeA = [&](int chunk) -> v16h {
        const int i  = chunk >> 2;
        const int j0 = (chunk & 3) * 32 + g * 16;
        const _Float16 aih = flsh[zr][i];
        v16h av;
        #pragma unroll
        for (int e = 0; e < 16; ++e) av[e] = aih;
        const v8h flo = *(const v8h*)&flsh[zr][j0];
        const v8h fhi = *(const v8h*)&flsh[zr][j0 + 8];
        const v16h fj = __builtin_shufflevector(
            flo, fhi, 0,1,2,3,4,5,6,7, 8,9,10,11,12,13,14,15);
        return av * fj;                      // 8x v_pk_mul_f16
    };

    v16h b0[4], b1[4], a0, a1;
    loadB(0, b0);
    a0 = makeA(0);
    // steady state: consume chunks 0..509, always preload chunk+1, chunk+2
    for (int chunk = 0; chunk < 510; chunk += 2) {
        loadB(chunk + 1, b1);
        a1 = makeA(chunk + 1);
        #pragma unroll
        for (int q = 0; q < 4; ++q)
            acc[q] = __builtin_amdgcn_wmma_f32_16x16x32_f16(
                         false, a0, false, b0[q], (short)0, acc[q], false, false);
        loadB(chunk + 2, b0);
        a0 = makeA(chunk + 2);
        #pragma unroll
        for (int q = 0; q < 4; ++q)
            acc[q] = __builtin_amdgcn_wmma_f32_16x16x32_f16(
                         false, a1, false, b1[q], (short)0, acc[q], false, false);
    }
    // epilogue: chunk 510 resident in b0/a0; load and do 511
    loadB(511, b1);
    a1 = makeA(511);
    #pragma unroll
    for (int q = 0; q < 4; ++q)
        acc[q] = __builtin_amdgcn_wmma_f32_16x16x32_f16(
                     false, a0, false, b0[q], (short)0, acc[q], false, false);
    #pragma unroll
    for (int q = 0; q < 4; ++q)
        acc[q] = __builtin_amdgcn_wmma_f32_16x16x32_f16(
                     false, a1, false, b1[q], (short)0, acc[q], false, false);

    #pragma unroll
    for (int q = 0; q < 4; ++q)
        #pragma unroll
        for (int e = 0; e < 8; ++e)
            out[(size_t)(z0 + e + g * 8) * K_OUT + kg * 64 + q * 16 + r]
                = acc[q][e];
}

// ---------------------------------------------------------------------------
extern "C" void kernel_launch(void* const* d_in, const int* in_sizes, int n_in,
                              void* d_out, int out_size, void* d_ws, size_t ws_size,
                              hipStream_t stream)
{
    (void)in_sizes; (void)n_in; (void)out_size; (void)ws_size;

    const float* feat = (const float*)d_in[0];   // [Z, 127]
    const float* W    = (const float*)d_in[1];   // [128, 2080]
    const float* T    = (const float*)d_in[2];   // [2080, 16384]
    float*       out  = (float*)d_out;           // [Z, 128]

    // workspace: P f16 [128,16384] = 4 MiB, then W16 f16 [128,2080] = 532 KB
    _Float16* P   = (_Float16*)d_ws;
    _Float16* W16 = (_Float16*)((char*)d_ws + (size_t)K_OUT * NN2 * sizeof(_Float16));

    cvt_w_f16<<<(K_OUT * DIM_TS + 255) / 256, 256, 0, stream>>>(W, W16);

    // Stage 1: 512 blocks x (2 c-tiles x 4 kc-groups) = 4096 waves.
    wt_gemm_f16<<<512, 256, 0, stream>>>(W16, T, P);

    // Stage 2: 256 blocks x (64 z x 128 k).
    tsq_gemm_f16<<<256, 256, 0, stream>>>(feat, P, out);
}
